// AttentionMechanism_14705968021847
// MI455X (gfx1250) — matrix operand order repaired and checked by reading, not compiled, not run on hardware
//
#include <hip/hip_runtime.h>
#include <hip/hip_bf16.h>
#include <math.h>

// Problem sizes (fixed by the reference)
#define ENC_DIM 512
#define ATT_DIM 256
#define BATCH   64
#define TLEN    4096
#define TILE_T  128   // rows of enc per workgroup in the WMMA kernel

#if __has_builtin(__builtin_amdgcn_global_load_async_to_lds_b128) && \
    __has_builtin(__builtin_amdgcn_s_wait_asynccnt)
#define HAVE_ASYNC_LDS 1
#else
#define HAVE_ASYNC_LDS 0
#endif

typedef __attribute__((ext_vector_type(16))) __bf16 v16bf;
typedef __attribute__((ext_vector_type(8)))  float  v8f;
typedef int v4i __attribute__((vector_size(16)));    // matches async builtin param type
typedef __attribute__((address_space(1))) v4i* v4i_gptr;
typedef __attribute__((address_space(3))) v4i* v4i_lptr;

// ---- fast tanh: CDNA5 has a hardware V_TANH_F32 transcendental ----
#if __has_builtin(__builtin_amdgcn_tanhf)
#define fast_tanh(x) __builtin_amdgcn_tanhf(x)
#elif __has_builtin(__builtin_amdgcn_tanh_f32)
#define fast_tanh(x) __builtin_amdgcn_tanh_f32(x)
#else
#define fast_tanh(x) tanhf(x)
#endif

static __device__ __forceinline__ unsigned int bf16_bits(float f) {
    unsigned int u = __float_as_uint(f);
    u += 0x7FFFu + ((u >> 16) & 1u);   // round-to-nearest-even
    return u >> 16;
}

// ---- pack two f32 -> packed bf16 pair; prefer v_cvt_pk_bf16_f32 ----
static __device__ __forceinline__ unsigned int pack_bf16(float lo, float hi) {
#if __has_builtin(__builtin_amdgcn_cvt_pk_bf16_f32)
    auto r = __builtin_amdgcn_cvt_pk_bf16_f32(lo, hi);
    unsigned int u;
    __builtin_memcpy(&u, &r, 4);
    return u;
#else
    return bf16_bits(lo) | (bf16_bits(hi) << 16);
#endif
}

struct Frag32B { uint4 lo; uint4 hi; };
static __device__ __forceinline__ v16bf make_frag(uint4 lo, uint4 hi) {
    union { Frag32B q; v16bf v; } u;
    u.q.lo = lo; u.q.hi = hi;
    return u.v;
}

// ---------------------------------------------------------------------------
// Kernel 0a: dec_proj[b][a] = decoder_hidden[b] . W_dec[:,a]      (64 x 256)
// ---------------------------------------------------------------------------
__global__ __launch_bounds__(256) void attn_dec_proj(
    const float* __restrict__ dh, const float* __restrict__ W_dec,
    float* __restrict__ dec_proj)
{
    const int b = blockIdx.x;
    const int a = threadIdx.x;
    float acc = 0.f;
    for (int k = 0; k < ENC_DIM; ++k)
        acc = fmaf(dh[(size_t)b * ENC_DIM + k], W_dec[(size_t)k * ATT_DIM + a], acc);
    dec_proj[(size_t)b * ATT_DIM + a] = acc;
}

// ---------------------------------------------------------------------------
// Kernel 0b: Wt_bf16[a][e] = bf16(W_enc[e][a])   (transpose + convert, once)
// ---------------------------------------------------------------------------
__global__ __launch_bounds__(256) void attn_conv_w(
    const float* __restrict__ W_enc, unsigned short* __restrict__ Wt)
{
    const int e = blockIdx.x;    // 0..511
    const int a = threadIdx.x;   // 0..255
    Wt[(size_t)a * ENC_DIM + e] = (unsigned short)(pack_bf16(W_enc[(size_t)e * ATT_DIM + a], 0.f) & 0xFFFFu);
}

// ---------------------------------------------------------------------------
// Kernel 1: fused enc_proj GEMM (bf16 WMMA) + tanh + dot(v_att) -> scores
// Block = 256 threads (8 wave32), owns (b, 128 t-rows). Each wave: 16 rows,
// full N=256 as 16 accumulator tiles of v_wmma_f32_16x16x32_bf16.
// K-loop: async-DMA W chunk to LDS while VALU packs enc f32->bf16; next
// K-chunk of enc is register-prefetched so global loads overlap the WMMAs.
// ---------------------------------------------------------------------------
__global__ __launch_bounds__(256) void attn_scores_wmma(
    const float* __restrict__ enc, const unsigned short* __restrict__ WtBf,
    const float* __restrict__ b_enc, const float* __restrict__ dec_proj,
    const float* __restrict__ v_att, const float* __restrict__ b_att,
    float* __restrict__ scores)
{
    __shared__ __align__(16) unsigned short sA[TILE_T * 32];   // 8 KB  (128 rows x 32 K, bf16)
    __shared__ __align__(16) unsigned short sW[ATT_DIM * 32];  // 16 KB (256 cols x 32 K, bf16)
    __shared__ float sBD[ATT_DIM];                             // b_enc + dec_proj[b]
    __shared__ float sV[ATT_DIM];                              // v_att

    const int tid   = threadIdx.x;
    const int b     = blockIdx.y;
    const int t0    = blockIdx.x * TILE_T;
    const int wave  = tid >> 5;
    const int lane  = tid & 31;
    const int lrow  = lane & 15;
    const int lhalf = lane >> 4;

    sBD[tid] = b_enc[tid] + dec_proj[(size_t)b * ATT_DIM + tid];
    sV[tid]  = v_att[tid];

    v8f acc[16] = {};

    // A-staging role: thread covers half a row (16 f32 -> 16 bf16 = 32B)
    const int arow  = tid >> 1;
    const int ahalf = tid & 1;
    const float* encRow = enc + ((size_t)(b * TLEN + t0 + arow)) * ENC_DIM + ahalf * 16;
    const unsigned short* wRow = WtBf + (size_t)tid * ENC_DIM;

    uint4* sA4 = (uint4*)sA;   // 4 uint4 per 32-element bf16 row
    uint4* sW4 = (uint4*)sW;

    // Register prefetch of the first enc K-chunk
    float4 f0, f1, f2, f3;
    {
        const float4* g = (const float4*)(encRow);
        f0 = g[0]; f1 = g[1]; f2 = g[2]; f3 = g[3];
    }

    for (int kc = 0; kc < ENC_DIM / 32; ++kc) {
        // ---- kick off async DMA of the W chunk (bf16, 64B/thread) first so
        //      it overlaps the f32->bf16 VALU packing below ----
#if HAVE_ASYNC_LDS
        {
            v4i_gptr gsrc = (v4i_gptr)(unsigned short*)(const_cast<unsigned short*>(wRow) + kc * 32);
            v4i_lptr ldst = (v4i_lptr)(sW + (size_t)tid * 32);
            __builtin_amdgcn_global_load_async_to_lds_b128(gsrc, ldst, 0, 0);
            __builtin_amdgcn_global_load_async_to_lds_b128(gsrc, ldst, 16, 0);
            __builtin_amdgcn_global_load_async_to_lds_b128(gsrc, ldst, 32, 0);
            __builtin_amdgcn_global_load_async_to_lds_b128(gsrc, ldst, 48, 0);
        }
#else
        {
            const uint4* s4 = (const uint4*)(wRow + kc * 32);
            sW4[tid * 4 + 0] = s4[0]; sW4[tid * 4 + 1] = s4[1];
            sW4[tid * 4 + 2] = s4[2]; sW4[tid * 4 + 3] = s4[3];
        }
#endif

        // ---- pack current enc chunk f32 -> bf16 into LDS ----
        uint4 p0, p1;
        p0.x = pack_bf16(f0.x, f0.y); p0.y = pack_bf16(f0.z, f0.w);
        p0.z = pack_bf16(f1.x, f1.y); p0.w = pack_bf16(f1.z, f1.w);
        p1.x = pack_bf16(f2.x, f2.y); p1.y = pack_bf16(f2.z, f2.w);
        p1.z = pack_bf16(f3.x, f3.y); p1.w = pack_bf16(f3.z, f3.w);
        sA4[arow * 4 + ahalf * 2 + 0] = p0;
        sA4[arow * 4 + ahalf * 2 + 1] = p1;

#if HAVE_ASYNC_LDS
        __builtin_amdgcn_s_wait_asynccnt(0);
#endif
        __syncthreads();

        // ---- register-prefetch next enc chunk; overlaps the WMMAs ----
        if (kc + 1 < ENC_DIM / 32) {
            const float4* g = (const float4*)(encRow + (kc + 1) * 32);
            f0 = g[0]; f1 = g[1]; f2 = g[2]; f3 = g[3];
        }

        // A fragment: 16x32 bf16, lanes 0-15 row M=lrow K0-7/K16-23 pattern
        v16bf afrag = make_frag(sA4[(wave * 16 + lrow) * 4 + lhalf],
                                sA4[(wave * 16 + lrow) * 4 + 2 + lhalf]);
#pragma unroll
        for (int nt = 0; nt < 16; ++nt) {
            // B fragment: 32x16 bf16, lane = column, K0-7/K16-23 pattern (same as A)
            v16bf bfrag = make_frag(sW4[(nt * 16 + lrow) * 4 + lhalf],
                                    sW4[(nt * 16 + lrow) * 4 + 2 + lhalf]);
            acc[nt] = __builtin_amdgcn_wmma_f32_16x16x32_bf16(
                false, afrag, false, bfrag, (short)0, acc[nt], false, false);
        }
        __syncthreads();
    }

    // Fused epilogue: score contribution = v_att[col] * tanh(acc + bias + dec)
    float part[8] = {0.f, 0.f, 0.f, 0.f, 0.f, 0.f, 0.f, 0.f};
#pragma unroll
    for (int nt = 0; nt < 16; ++nt) {
        const int col   = nt * 16 + lrow;
        const float bias = sBD[col];
        const float va   = sV[col];
#pragma unroll
        for (int r = 0; r < 8; ++r)
            part[r] = fmaf(va, fast_tanh(acc[nt][r] + bias), part[r]);
    }

    const float battv = b_att[0];
#pragma unroll
    for (int r = 0; r < 8; ++r) {
        float v = part[r];
        v += __shfl_xor(v, 1, 32);
        v += __shfl_xor(v, 2, 32);
        v += __shfl_xor(v, 4, 32);
        v += __shfl_xor(v, 8, 32);
        if (lrow == 0) {
            // C/D layout: VGPR r holds rows r (lanes 0-15) and r+8 (lanes 16-31)
            const int trow = t0 + wave * 16 + r + (lhalf << 3);
            scores[(size_t)b * TLEN + trow] = v + battv;
        }
    }
}

// ---------------------------------------------------------------------------
// Kernel 2: row softmax over T=4096 -> attention weights (output 2)
// ---------------------------------------------------------------------------
__global__ __launch_bounds__(256) void attn_softmax(
    const float* __restrict__ scores, float* __restrict__ out_w)
{
    const int b = blockIdx.x;
    const int tid = threadIdx.x;
    __shared__ float red[256];
    const float* s = scores + (size_t)b * TLEN;

    float m = -3.402823466e38f;
    for (int i = tid; i < TLEN; i += 256) m = fmaxf(m, s[i]);
    red[tid] = m; __syncthreads();
    for (int o = 128; o > 0; o >>= 1) {
        if (tid < o) red[tid] = fmaxf(red[tid], red[tid + o]);
        __syncthreads();
    }
    m = red[0]; __syncthreads();

    float sum = 0.f;
    for (int i = tid; i < TLEN; i += 256) sum += __expf(s[i] - m);
    red[tid] = sum; __syncthreads();
    for (int o = 128; o > 0; o >>= 1) {
        if (tid < o) red[tid] += red[tid + o];
        __syncthreads();
    }
    const float inv = 1.0f / red[0];

    for (int i = tid; i < TLEN; i += 256)
        out_w[(size_t)b * TLEN + i] = __expf(s[i] - m) * inv;
}

// ---------------------------------------------------------------------------
// Kernel 3: context[b][e] = sum_t w[b][t] * enc[b][t][e]  (memory-bound pass)
// ---------------------------------------------------------------------------
__global__ __launch_bounds__(128) void attn_context(
    const float* __restrict__ enc, const float* __restrict__ w,
    float* __restrict__ ctx)
{
    const int tid = threadIdx.x;
    const int b  = blockIdx.y;
    const int e0 = blockIdx.x * 128;
    __shared__ float swt[128];

    float acc = 0.f;
    for (int t = 0; t < TLEN; t += 128) {
        __syncthreads();
        swt[tid] = w[(size_t)b * TLEN + t + tid];
        __syncthreads();
#pragma unroll 8
        for (int j = 0; j < 128; ++j)
            acc = fmaf(swt[j], enc[((size_t)(b * TLEN + t + j)) * ENC_DIM + e0 + tid], acc);
    }
    ctx[(size_t)b * ENC_DIM + e0 + tid] = acc;
}

// ---------------------------------------------------------------------------
// Host-side launcher
// ---------------------------------------------------------------------------
extern "C" void kernel_launch(void* const* d_in, const int* in_sizes, int n_in,
                              void* d_out, int out_size, void* d_ws, size_t ws_size,
                              hipStream_t stream)
{
    (void)in_sizes; (void)n_in; (void)out_size; (void)ws_size;

    const float* enc   = (const float*)d_in[0];  // (B,T,E)
    const float* dh    = (const float*)d_in[1];  // (B,DEC)
    const float* W_enc = (const float*)d_in[2];  // (E,A)
    const float* b_enc = (const float*)d_in[3];  // (A)
    const float* W_dec = (const float*)d_in[4];  // (DEC,A)
    const float* v_att = (const float*)d_in[5];  // (A)
    const float* b_att = (const float*)d_in[6];  // scalar

    float* out_ctx = (float*)d_out;                          // (B,E) first
    float* out_w   = out_ctx + (size_t)BATCH * ENC_DIM;      // (B,T) second

    char* ws = (char*)d_ws;
    float*          dec_proj = (float*)ws;                              // 64 KB
    unsigned short* WtBf     = (unsigned short*)(ws + 64 * 1024);       // 256 KB
    float*          scores   = (float*)(ws + 64 * 1024 + 256 * 1024);   // 1 MB

    attn_dec_proj<<<dim3(BATCH), 256, 0, stream>>>(dh, W_dec, dec_proj);
    attn_conv_w <<<dim3(ENC_DIM), 256, 0, stream>>>(W_enc, WtBf);
    attn_scores_wmma<<<dim3(TLEN / TILE_T, BATCH), 256, 0, stream>>>(
        enc, WtBf, b_enc, dec_proj, v_att, b_att, scores);
    attn_softmax<<<dim3(BATCH), 256, 0, stream>>>(scores, out_w);
    attn_context<<<dim3(ENC_DIM / 128, BATCH), 128, 0, stream>>>(enc, out_w, out_ctx);
}
